// MHCACLayer_10282151707237
// MI455X (gfx1250) — matrix operand order repaired and checked by reading, not compiled, not run on hardware
//
#include <hip/hip_runtime.h>
#include <hip/hip_bf16.h>

typedef unsigned short u16;
typedef __attribute__((ext_vector_type(16))) __bf16 v16bf;
typedef __attribute__((ext_vector_type(8)))  float  v8f;

struct U4s { unsigned x, y, z, w; };
struct U8s { U4s lo, hi; };
struct U2s { unsigned x, y; };

static __device__ __forceinline__ u16 f2bf(float f) {
    return __builtin_bit_cast(u16, (__bf16)f);   // native v_cvt (RTNE)
}
static __device__ __forceinline__ float bf2f(u16 h) {
    return __uint_as_float(((unsigned)h) << 16);
}
static __device__ __forceinline__ v16bf frag_from(const u16* p0, const u16* p1) {
    U8s t;
    t.lo = *(const U4s*)p0;   // 16B contiguous chunk
    t.hi = *(const U4s*)p1;   // 16B contiguous chunk
    return __builtin_bit_cast(v16bf, t);
}

// ---------------------------------------------------------------------------
// Q projection: Q[a,q] = sum_d cq[a,d] * WQ[a,d,q] + bQ[a,q]   (14 x 64)
// ---------------------------------------------------------------------------
__global__ void q_kernel(const float* __restrict__ cq, const float* __restrict__ WQ,
                         const float* __restrict__ bQ, float* __restrict__ Qo) {
    int t = threadIdx.x;              // 896 threads
    int a = t >> 6, q = t & 63;
    float acc = bQ[t];
    const float* cqa = &cq[a * 64];
    const float* w   = &WQ[a * 4096 + q];
    #pragma unroll 8
    for (int d = 0; d < 64; ++d) acc += cqa[d] * w[d * 64];
    Qo[t] = acc;
}

// ---------------------------------------------------------------------------
// bf16 WMMA GEMM: C[128x128 tile] = A(M x 768) * B(768 x N)
//   MODE 0: B from {WK,WV} ([12][768][64] each, N=1536), +bias, bf16 out (K|V)
//   MODE 1: B = W1 row-major, +b1, ReLU, bf16 out (h)
//   MODE 2: B = W2 row-major, +b2, +x residual, f32 out (y)
// A is f32 (ABF16=0) or bf16 (ABF16=1), row stride lda.
// Software pipelined: tile kt+1 fetched into registers during compute of kt.
// ---------------------------------------------------------------------------
template <int MODE, int ABF16>
__global__ __launch_bounds__(256) void gemm_kernel(
    const void* __restrict__ Aptr, int lda,
    const float* __restrict__ B0, const float* __restrict__ B1,
    const float* __restrict__ bias0, const float* __restrict__ bias1,
    const float* __restrict__ xres,
    void* __restrict__ Cptr, int ldc)
{
    __shared__ __align__(16) u16 sA[128 * 40];   // [m][k], pad 40
    __shared__ __align__(16) u16 sB[128 * 40];   // [n][k] (transposed), pad 40

    const int t     = threadIdx.x;
    const int mbase = blockIdx.y * 128;
    const int nbase = blockIdx.x * 128;

    const int lane = t & 31;
    const int wv   = t >> 5;
    const int wm   = (wv & 3) * 32;   // wave M offset in tile
    const int wn   = (wv >> 2) * 64;  // wave N offset in tile
    const int lr   = lane & 15;
    const int lh   = lane >> 4;

    v8f acc[2][4];
    const v8f vzero = {0.f, 0.f, 0.f, 0.f, 0.f, 0.f, 0.f, 0.f};
    #pragma unroll
    for (int i = 0; i < 2; ++i)
        #pragma unroll
        for (int j = 0; j < 4; ++j) acc[i][j] = vzero;

    // loader assignments
    const int ar = t >> 3;            // A row 0..31 (+p*32)
    const int ac = (t & 7) * 4;       // A k   0..28
    const int bn = (t & 31) * 4;      // B n   0..124
    const int bk = t >> 5;            // B k   0..7 (+p*8)

    float4 pa[4];                     // prefetched A (f32 path)
    U2s    pab[4];                    // prefetched A (bf16 path)
    float4 pb[4];                     // prefetched B

    auto fetch = [&](int kt) {
        const int kb = kt * 32;
        #pragma unroll
        for (int p = 0; p < 4; ++p) {
            int row = ar + p * 32;
            size_t gi = (size_t)(mbase + row) * lda + (kb + ac);
            if constexpr (ABF16) pab[p] = *(const U2s*)((const u16*)Aptr + gi);
            else                 pa[p]  = *(const float4*)((const float*)Aptr + gi);
        }
        #pragma unroll
        for (int p = 0; p < 4; ++p) {
            int c = kb + bk + p * 8;
            if constexpr (MODE == 0) {
                int gn = nbase + bn;
                const float* Wp = (gn < 768) ? B0 : B1;
                int nn = gn & 767;   // W[h][c][q]; q-run of 4 stays inside one head block
                pb[p] = *(const float4*)&Wp[(size_t)((nn >> 6) * 768 + c) * 64 + (nn & 63)];
            } else {
                pb[p] = *(const float4*)&B0[(size_t)c * 768 + (nbase + bn)];
            }
        }
    };

    auto stage = [&]() {
        #pragma unroll
        for (int p = 0; p < 4; ++p) {
            int row = ar + p * 32;
            U2s pk;
            if constexpr (ABF16) {
                pk = pab[p];
            } else {
                pk.x = (unsigned)f2bf(pa[p].x) | ((unsigned)f2bf(pa[p].y) << 16);
                pk.y = (unsigned)f2bf(pa[p].z) | ((unsigned)f2bf(pa[p].w) << 16);
            }
            *(U2s*)&sA[row * 40 + ac] = pk;
        }
        #pragma unroll
        for (int p = 0; p < 4; ++p) {
            int k = bk + p * 8;
            sB[(bn + 0) * 40 + k] = f2bf(pb[p].x);
            sB[(bn + 1) * 40 + k] = f2bf(pb[p].y);
            sB[(bn + 2) * 40 + k] = f2bf(pb[p].z);
            sB[(bn + 3) * 40 + k] = f2bf(pb[p].w);
        }
    };

    fetch(0);
    for (int kt = 0; kt < 24; ++kt) {
        __syncthreads();          // previous compute done, LDS reusable
        stage();
        __syncthreads();          // tile visible
        if (kt < 23) fetch(kt + 1);   // overlap next global fetch with compute

        // ---- fragments per documented gfx1250 layouts ----
        v16bf af[2], bfg[4];
        #pragma unroll
        for (int fm = 0; fm < 2; ++fm) {
            const u16* p = &sA[(wm + fm * 16 + lr) * 40 + lh * 8];
            af[fm] = frag_from(p, p + 16);   // K {lh*8..+7} and {16+lh*8..+7}
        }
        #pragma unroll
        for (int fn = 0; fn < 4; ++fn) {
            const u16* p = &sB[(wn + fn * 16 + lr) * 40 + lh * 8];
            bfg[fn] = frag_from(p, p + 16);   // K {lh*8..+7} and {16+lh*8..+7} at N=lr
        }
        #pragma unroll
        for (int fm = 0; fm < 2; ++fm)
            #pragma unroll
            for (int fn = 0; fn < 4; ++fn)
                acc[fm][fn] = __builtin_amdgcn_wmma_f32_16x16x32_bf16(
                    false, af[fm], false, bfg[fn], (short)0, acc[fm][fn], false, false);
    }

    // ---- epilogue ----
    #pragma unroll
    for (int fm = 0; fm < 2; ++fm) {
        #pragma unroll
        for (int fn = 0; fn < 4; ++fn) {
            int gcol = nbase + wn + fn * 16 + lr;
            float bs;
            if constexpr (MODE == 0) bs = (gcol < 768) ? bias0[gcol] : bias1[gcol - 768];
            else                     bs = bias0[gcol];
            int grow0 = mbase + wm + fm * 16 + lh * 8;
            #pragma unroll
            for (int r = 0; r < 8; ++r) {
                int grow = grow0 + r;
                float v = acc[fm][fn][r] + bs;
                size_t oi = (size_t)grow * ldc + gcol;
                if constexpr (MODE == 0) {
                    ((u16*)Cptr)[oi] = f2bf(v);
                } else if constexpr (MODE == 1) {
                    v = v > 0.f ? v : 0.f;
                    ((u16*)Cptr)[oi] = f2bf(v);
                } else {
                    v += xres[(size_t)grow * 768 + gcol];
                    ((float*)Cptr)[oi] = v;
                }
            }
        }
    }
}

// ---------------------------------------------------------------------------
// Per-batch attention: scores = Q*K^T/8, softmax over query axis 'a',
// out = attn*V (heads concat), write cls + out (pre-LN) to Xo.
// One block per batch item b. KV rows are bf16 [14][1536] (K | V).
// ---------------------------------------------------------------------------
__global__ __launch_bounds__(256) void attn_kernel(
    const float* __restrict__ Q, const u16* __restrict__ KV,
    const float* __restrict__ cls, float* __restrict__ Xo)
{
    __shared__ float sQ[14 * 64];
    __shared__ u16   sKV[14 * 1536];
    __shared__ float sS[12 * 14 * 14];
    const int t = threadIdx.x;
    const int b = blockIdx.x;

    for (int i = t; i < 896; i += 256) sQ[i] = Q[i];
    {
        const unsigned* src = (const unsigned*)(KV + (size_t)b * 21504);
        unsigned* dst = (unsigned*)sKV;
        for (int i = t; i < 10752; i += 256) dst[i] = src[i];
    }
    __syncthreads();

    // scores[h][a][s]
    for (int i = t; i < 2352; i += 256) {
        int h = i / 196, r = i % 196, a = r / 14, s = r % 14;
        const float* q = &sQ[a * 64];
        const u16*   k = &sKV[s * 1536 + h * 64];
        float acc = 0.f;
        #pragma unroll 8
        for (int j = 0; j < 64; ++j) acc += q[j] * bf2f(k[j]);
        sS[h * 196 + a * 14 + s] = acc * 0.125f;
    }
    __syncthreads();

    // softmax over 'a' for each (h, s)
    if (t < 168) {
        int h = t / 14, s = t % 14;
        float* p = &sS[h * 196 + s];
        float mx = -3.4e38f;
        #pragma unroll
        for (int a = 0; a < 14; ++a) mx = fmaxf(mx, p[a * 14]);
        float e[14], sum = 0.f;
        #pragma unroll
        for (int a = 0; a < 14; ++a) { e[a] = __expf(p[a * 14] - mx); sum += e[a]; }
        float inv = 1.f / sum;
        #pragma unroll
        for (int a = 0; a < 14; ++a) p[a * 14] = e[a] * inv;
    }
    __syncthreads();

    // out[a][c] = sum_s attn[h][a][s] * V[s][c];  Xo = cls + out (pre-LN)
    for (int i = t; i < 10752; i += 256) {
        int a = i / 768, c = i % 768, h = c >> 6;
        float acc = 0.f;
        #pragma unroll
        for (int s = 0; s < 14; ++s)
            acc += sS[h * 196 + a * 14 + s] * bf2f(sKV[s * 1536 + 768 + c]);
        size_t gi = ((size_t)b * 14 + a) * 768 + c;
        Xo[gi] = cls[gi] + acc;
    }
}

// ---------------------------------------------------------------------------
// Row layernorm over 768, one wave32 per row, 8 rows per block.
// ---------------------------------------------------------------------------
__global__ __launch_bounds__(256) void ln_kernel(
    const float* __restrict__ in, float* __restrict__ out,
    const float* __restrict__ g, const float* __restrict__ bb)
{
    const int lane = threadIdx.x & 31;
    const int w    = threadIdx.x >> 5;
    const size_t row = (size_t)blockIdx.x * 8 + w;
    const float* p = in + row * 768;
    float e[24], s = 0.f;
    #pragma unroll
    for (int j = 0; j < 24; ++j) { e[j] = p[lane + j * 32]; s += e[j]; }
    #pragma unroll
    for (int m = 16; m >= 1; m >>= 1) s += __shfl_xor(s, m, 32);
    float mean = s * (1.f / 768.f);
    float v = 0.f;
    #pragma unroll
    for (int j = 0; j < 24; ++j) { float d = e[j] - mean; v += d * d; }
    #pragma unroll
    for (int m = 16; m >= 1; m >>= 1) v += __shfl_xor(v, m, 32);
    float rstd = rsqrtf(v * (1.f / 768.f) + 1e-5f);
    float* o = out + row * 768;
    #pragma unroll
    for (int j = 0; j < 24; ++j) {
        int c = lane + j * 32;
        o[c] = (e[j] - mean) * rstd * g[c] + bb[c];
    }
}

// ---------------------------------------------------------------------------
extern "C" void kernel_launch(void* const* d_in, const int* in_sizes, int n_in,
                              void* d_out, int out_size, void* d_ws, size_t ws_size,
                              hipStream_t stream) {
    (void)in_sizes; (void)n_in; (void)out_size; (void)ws_size;
    const float* cq  = (const float*)d_in[0];
    const float* cls = (const float*)d_in[1];
    const float* WQ  = (const float*)d_in[2];
    const float* bQ  = (const float*)d_in[3];
    const float* WK  = (const float*)d_in[4];
    const float* bK  = (const float*)d_in[5];
    const float* WV  = (const float*)d_in[6];
    const float* bV  = (const float*)d_in[7];
    const float* W1  = (const float*)d_in[8];
    const float* b1  = (const float*)d_in[9];
    const float* W2  = (const float*)d_in[10];
    const float* b2  = (const float*)d_in[11];
    const float* g1  = (const float*)d_in[12];
    const float* be1 = (const float*)d_in[13];
    const float* g2  = (const float*)d_in[14];
    const float* be2 = (const float*)d_in[15];

    char* wsb = (char*)d_ws;
    const size_t SZ = 176160768ull;               // 57344*768*4 == 57344*1536*2
    float* Qw = (float*)wsb;                      // 3.5 KB (reserve 4 KB)
    u16*   KV = (u16*)(wsb + 4096);               // bf16 K|V, SZ bytes
    float* Y  = (float*)(wsb + 4096);             // reuses KV region (KV dead by then)
    float* X  = (float*)(wsb + 4096 + SZ);        // f32 x, SZ bytes
    u16*   Hb = (u16*)(wsb + 4096 + 2 * SZ);      // bf16 hidden, SZ/2 bytes

    // 1. Q projection
    q_kernel<<<1, 896, 0, stream>>>(cq, WQ, bQ, Qw);
    // 2. K & V projections (fused, N=1536), bf16 out with bias
    gemm_kernel<0, 0><<<dim3(12, 448), 256, 0, stream>>>(
        (const void*)cls, 768, WK, WV, bK, bV, nullptr, (void*)KV, 1536);
    // 3. attention + residual (pre-LN) -> X
    attn_kernel<<<4096, 256, 0, stream>>>(Qw, KV, cls, X);
    // 4. LN1 in place
    ln_kernel<<<7168, 256, 0, stream>>>(X, X, g1, be1);
    // 5. h = relu(x*W1 + b1), bf16 out
    gemm_kernel<1, 0><<<dim3(6, 448), 256, 0, stream>>>(
        (const void*)X, 768, W1, nullptr, b1, nullptr, nullptr, (void*)Hb, 768);
    // 6. y = h*W2 + b2 + x, f32 out (overwrites KV region)
    gemm_kernel<2, 1><<<dim3(6, 448), 256, 0, stream>>>(
        (const void*)Hb, 768, W2, nullptr, b2, nullptr, X, (void*)Y, 768);
    // 7. LN2 -> output
    ln_kernel<<<7168, 256, 0, stream>>>(Y, (float*)d_out, g2, be2);
}
